// ScreeningUnit_22840636080386
// MI455X (gfx1250) — matrix operand order repaired and checked by reading, not compiled, not run on hardware
//
#include <hip/hip_runtime.h>

// MI455X / gfx1250, wave32. f16 WMMA (v_wmma_f32_16x16x32_f16) with f32 accum.
// Flash-style screened attention: one WG = 64 q-rows of one (b,h); 4 waves x 16 rows.

typedef __attribute__((ext_vector_type(16))) _Float16 v16h;
typedef __attribute__((ext_vector_type(8)))  _Float16 v8h;
typedef __attribute__((ext_vector_type(8)))  float    v8f;

#define SEQ   1024
#define DIM   64
#define QTILE 64
#define KTILE 64
#define LPADH 72   // padded row stride in halves (bank stride 36 -> conflict-free over 16 lanes)
#define EPSF  1e-6f

static __device__ __forceinline__ v16h join16(v8h lo, v8h hi) {
  return __builtin_shufflevector(lo, hi, 0,1,2,3,4,5,6,7,8,9,10,11,12,13,14,15);
}

__global__ __launch_bounds__(128) void screening_unit_kernel(
    const float* __restrict__ Qg, const float* __restrict__ Kg,
    const float* __restrict__ Vg, const float* __restrict__ p_sw,
    const float* __restrict__ p_sr, float* __restrict__ out)
{
  __shared__ __align__(16) _Float16 Qs[64][LPADH];      // q tile, unit+rope, f16 [row][d]
  __shared__ __align__(16) _Float16 Ks[64][LPADH];      // key block, unit+rope, [key][d]
  __shared__ __align__(16) _Float16 Vt[64][LPADH];      // value block, unit, TRANSPOSED [d][key]
  __shared__ __align__(16) _Float16 Ws[4][16][LPADH];   // per-wave screened weights [m][key]

  const int tid  = threadIdx.x;
  const int lane = tid & 31;
  const int wv   = tid >> 5;                 // wave id 0..3 (wave32)
  const int ntiles = SEQ / QTILE;            // 16
  const int bh   = blockIdx.x / ntiles;
  const int tile = blockIdx.x % ntiles;
  const int q0   = tile * QTILE;

  // ---- scalars (window / relevance) ----
  const float s_w = p_sw[0], s_r = p_sr[0];
  const float window = (s_w > 20.f) ? s_w : log1pf(expf(s_w));   // softplus
  const float r      = 1.f / (1.f + expf(-s_r));                 // sigmoid
  const float inv_r  = 1.f / r;
  const float trim   = 1.f - r;
  const float inv_win = (window > 1024.f) ? 0.f : (1.f / fmaxf(window, EPSF)); // MiPE off past threshold

  const size_t bhoff = (size_t)bh * SEQ * DIM;
  const float* qb = Qg + bhoff;
  const float* kb = Kg + bhoff;
  const float* vb = Vg + bhoff;

  // ---- direct row load (L2-resident; 16 x b128), returns sumsq ----
  float xr[64];
  auto load_row = [&](const float* src) {
    float ss = 0.f;
#pragma unroll
    for (int c4 = 0; c4 < 16; ++c4) {
      const float4 f = *(const float4*)(src + 4 * c4);
      xr[4*c4+0] = f.x; xr[4*c4+1] = f.y; xr[4*c4+2] = f.z; xr[4*c4+3] = f.w;
      ss += f.x*f.x + f.y*f.y + f.z*f.z + f.w*f.w;
    }
    return ss;
  };

  // ---- stage Q once: unit-normalize + RoPE, one row per thread (tids 0..63) ----
  if (tid < 64) {
    const int pos = q0 + tid;
    const float ss  = load_row(qb + (size_t)pos * DIM);
    const float inv = 1.f / (sqrtf(ss) + EPSF);
    const float th  = (float)pos * inv_win;
    const float c = cosf(th), s = sinf(th);
#pragma unroll
    for (int d = 0; d < 32; ++d) {
      float x1 = xr[d] * inv, x2 = xr[d + 32] * inv;
      Qs[tid][d]      = (_Float16)(x1 * c - x2 * s);
      Qs[tid][d + 32] = (_Float16)(x1 * s + x2 * c);
    }
  }
  __syncthreads();

  // Persistent A-fragments for Q (ISA 16-bit A 16x32 layout).
  const int n    = lane & 15;                 // C/B column within 16-tile
  const int moff = (lane < 16) ? 0 : 8;       // C row offset
  const int kyb  = (lane < 16) ? 0 : 8;       // B contraction base
  const int am   = wv * 16 + n;               // A row for this lane
  const int ahb  = (lane < 16) ? 0 : 8;       // A contraction base
  v16h aQ[2];
#pragma unroll
  for (int c = 0; c < 2; ++c) {
    v16h a;
#pragma unroll
    for (int g = 0; g < 4; ++g) {
      int f1 = 32 * c + ahb + 2 * g;
      a[2*g]     = Qs[am][f1];     a[2*g+1]   = Qs[am][f1+1];
      int f2 = f1 + 16;
      a[8+2*g]   = Qs[am][f2];     a[8+2*g+1] = Qs[am][f2+1];
    }
    aQ[c] = a;
  }

  v8f acc[4];
#pragma unroll
  for (int c = 0; c < 4; ++c) acc[c] = (v8f){0.f,0.f,0.f,0.f,0.f,0.f,0.f,0.f};

  // ---- causal loop over 64-key blocks ----
  for (int blk = 0; blk <= tile; ++blk) {
    __syncthreads();   // previous compute done before Ks/Vt rewrite

    // Stage: tids 0..63 -> K row (norm+rope, row-major); tids 64..127 -> V row (norm, transposed)
    {
      const int rrow = tid & 63;
      const int pos  = blk * KTILE + rrow;
      const float* src = ((tid < 64) ? kb : vb) + (size_t)pos * DIM;
      const float ss  = load_row(src);
      const float inv = 1.f / (sqrtf(ss) + EPSF);
      if (tid < 64) {
        const float th = (float)pos * inv_win;
        const float c = cosf(th), s = sinf(th);
#pragma unroll
        for (int d = 0; d < 32; ++d) {
          float x1 = xr[d] * inv, x2 = xr[d + 32] * inv;
          Ks[rrow][d]      = (_Float16)(x1 * c - x2 * s);
          Ks[rrow][d + 32] = (_Float16)(x1 * s + x2 * c);
        }
      } else {
#pragma unroll
        for (int d = 0; d < 64; ++d)
          Vt[d][rrow] = (_Float16)(xr[d] * inv);
      }
    }

    // Prefetch next key block (global_prefetch_b8): 128 threads x 128B cover 16KB each of K,V
    if (blk < tile) {
      const size_t nxt = (size_t)(blk + 1) * KTILE * DIM + (size_t)tid * 32;
      __builtin_prefetch(kb + nxt, 0, 0);
      __builtin_prefetch(vb + nxt, 0, 0);
    }
    __syncthreads();

    const int qrow0 = q0 + wv * 16 + moff;    // global q row for C element i=0

    // S = Q K^T per 16-key tile, screen, emit f16 W to per-wave LDS
#pragma unroll
    for (int j = 0; j < 4; ++j) {
      v8f st = (v8f){0.f,0.f,0.f,0.f,0.f,0.f,0.f,0.f};
      const int key = j * 16 + n;
#pragma unroll
      for (int c = 0; c < 2; ++c) {
        const v8h* p = (const v8h*)&Ks[key][kyb + 32 * c];
        v16h b = join16(p[0], p[2]);
        st = __builtin_amdgcn_wmma_f32_16x16x32_f16(false, aQ[c], false, b,
                                                    (short)0, st, false, false);
      }
      const int key_g = blk * KTILE + key;
#pragma unroll
      for (int i = 0; i < 8; ++i) {
        float dist = (float)(qrow0 + i - key_g);
        float mask = (dist < 0.f) ? 0.f : 1.f / (1.f + expf(dist - window));
        float sim  = fminf(fmaxf(st[i], -1.f), 1.f);
        float rl   = fmaxf(sim - trim, 0.f) * inv_r;
        Ws[wv][moff + i][j * 16 + n] = (_Float16)(rl * rl * mask);
      }
    }

    // agg += W V over two 32-key sub-blocks (wave-private Ws: in-order LDS, no barrier)
#pragma unroll
    for (int sb = 0; sb < 2; ++sb) {
      v16h a2;
      const int kb0 = sb * 32 + ahb;
#pragma unroll
      for (int g = 0; g < 4; ++g) {
        a2[2*g]     = Ws[wv][n][kb0 + 2*g];      a2[2*g+1]   = Ws[wv][n][kb0 + 2*g + 1];
        a2[8+2*g]   = Ws[wv][n][kb0 + 16 + 2*g]; a2[8+2*g+1] = Ws[wv][n][kb0 + 17 + 2*g];
      }
#pragma unroll
      for (int c = 0; c < 4; ++c) {
        const int d = c * 16 + n;
        const v8h* p = (const v8h*)&Vt[d][sb * 32 + kyb];
        v16h b = join16(p[0], p[2]);
        acc[c] = __builtin_amdgcn_wmma_f32_16x16x32_f16(false, a2, false, b,
                                                        (short)0, acc[c], false, false);
      }
    }
  }

  // ---- tanh-norm epilogue: row norms via xor-shuffles within 16-lane half ----
  float scale[8];
#pragma unroll
  for (int i = 0; i < 8; ++i) {
    float ssq = 0.f;
#pragma unroll
    for (int c = 0; c < 4; ++c) ssq += acc[c][i] * acc[c][i];
#pragma unroll
    for (int m = 1; m < 16; m <<= 1) ssq += __shfl_xor(ssq, m, 32);
    float nrm = sqrtf(ssq);
    scale[i] = tanhf(nrm) / (nrm + EPSF);
  }
  float* orow = out + bhoff + (size_t)(q0 + wv * 16) * DIM;
#pragma unroll
  for (int c = 0; c < 4; ++c)
#pragma unroll
    for (int i = 0; i < 8; ++i)
      orow[(size_t)(moff + i) * DIM + c * 16 + n] = acc[c][i] * scale[i];
}

extern "C" void kernel_launch(void* const* d_in, const int* in_sizes, int n_in,
                              void* d_out, int out_size, void* d_ws, size_t ws_size,
                              hipStream_t stream) {
  (void)n_in; (void)out_size; (void)d_ws; (void)ws_size;
  const float* q  = (const float*)d_in[0];
  const float* k  = (const float*)d_in[1];
  const float* v  = (const float*)d_in[2];
  const float* sw = (const float*)d_in[3];
  const float* sr = (const float*)d_in[4];
  float* out = (float*)d_out;
  const int bh = in_sizes[0] / (SEQ * DIM);       // B*H = 32
  dim3 grid(bh * (SEQ / QTILE));                  // 512 workgroups
  hipLaunchKernelGGL(screening_unit_kernel, grid, dim3(128), 0, stream,
                     q, k, v, sw, sr, out);
}
